// DecoderModel_44427141709877
// MI455X (gfx1250) — matrix operand order repaired and checked by reading, not compiled, not run on hardware
//
#include <hip/hip_runtime.h>
#include <cstddef>

// ---------------- types ----------------
typedef __bf16 v16bf __attribute__((ext_vector_type(16)));
typedef float  v8f   __attribute__((ext_vector_type(8)));

struct __align__(16) U4 { unsigned a, b, c, d; };
struct U8 { U4 lo, hi; };

// ---------------- helpers ----------------
__device__ __forceinline__ unsigned short f2bf(float f) {
  unsigned u = __builtin_bit_cast(unsigned, f);
  u += 0x7FFFu + ((u >> 16) & 1u);            // round-to-nearest-even
  return (unsigned short)(u >> 16);
}
__device__ __forceinline__ float sigm(float x) { return 1.f / (1.f + __expf(-x)); }
__device__ __forceinline__ float tanh_fast(float x) {
  x = fminf(fmaxf(x, -15.f), 15.f);
  float e = __expf(2.f * x);
  return (e - 1.f) / (e + 1.f);
}

#define B_SZ 512
#define T_SZ 128
#define E_SZ 64
#define H_SZ 512
#define V_SZ 100

// ---------------- prologue kernels ----------------

// Tiled transpose fp32[K][N] -> bf16[N][K] (coalesced both sides via LDS)
__global__ __launch_bounds__(256) void pack_wT(const float* __restrict__ in,
                                               unsigned short* __restrict__ out,
                                               int K, int N) {
  __shared__ float tile[32][33];
  const int kb = blockIdx.x * 32, nb = blockIdx.y * 32;
  const int tx = threadIdx.x & 31, ty = threadIdx.x >> 5;   // ty 0..7
  for (int i = ty; i < 32; i += 8)
    tile[i][tx] = in[(size_t)(kb + i) * N + nb + tx];
  __syncthreads();
  for (int i = ty; i < 32; i += 8)
    out[(size_t)(nb + i) * K + kb + tx] = f2bf(tile[tx][i]);
}

// Wp fp32[512][100] -> bf16[112][512], zero-padded columns 100..111
__global__ __launch_bounds__(256) void pack_wp(const float* __restrict__ Wp,
                                               unsigned short* __restrict__ out) {
  int idx = blockIdx.x * 256 + threadIdx.x;
  if (idx >= 112 * 512) return;
  int n = idx >> 9, k = idx & 511;
  out[idx] = (n < V_SZ) ? f2bf(Wp[(size_t)k * V_SZ + n]) : (unsigned short)0;
}

// X[t][b][0:576] = [ emb(tok[b][t]) (64) | state[b] (512) ]  in bf16
__global__ __launch_bounds__(256) void build_x(const int* __restrict__ dib,
                                               const float* __restrict__ emb,
                                               const float* __restrict__ state,
                                               unsigned short* __restrict__ X) {
  size_t idx = (size_t)blockIdx.x * 256 + threadIdx.x;
  if (idx >= (size_t)T_SZ * B_SZ * 576) return;
  int k  = (int)(idx % 576);
  int bt = (int)(idx / 576);
  int t = bt >> 9, b = bt & 511;
  float v;
  if (k < E_SZ) {
    int tok = dib[(size_t)b * T_SZ + t];
    v = emb[(size_t)tok * E_SZ + k];
  } else {
    v = state[(size_t)b * H_SZ + (k - E_SZ)];
  }
  X[idx] = f2bf(v);
}

__global__ __launch_bounds__(256) void zero32(unsigned* __restrict__ p, int n) {
  int i = blockIdx.x * 256 + threadIdx.x;
  if (i < n) p[i] = 0u;
}

// ---------------- fused GEMM + LSTM cell step kernel ----------------
// Block: 128 threads (4 waves). Block tile: 64 batch rows x 32 hidden units
// (=> 128 gate columns: u, u+512, u+1024, u+1536). grid = (16 strips, 8 row-blocks).
// A = [A0 (KA0 cols, stride sA0) | A1 (KA1 cols, stride 512)] bf16 per batch row.
// W = bf16 [2048][KA0+KA1] (pre-transposed).
template <int KA0, int KA1>
__global__ __launch_bounds__(128) void lstm_step_kernel(
    const unsigned short* __restrict__ A0, int sA0,
    const unsigned short* __restrict__ A1,          // prev h state (also old-h source)
    const unsigned short* __restrict__ W,
    const float* __restrict__ bias,                 // [2048] (i|j|f|o)
    float* __restrict__ cbuf,                       // [512][512]
    unsigned short* __restrict__ hraw,              // [512][512] or nullptr
    unsigned short* __restrict__ hstate,            // next h state [512][512]
    unsigned short* __restrict__ outs_t,            // [512][512] or nullptr
    const int* __restrict__ lens, int t) {
  constexpr int KT  = KA0 + KA1;
  constexpr int NCH = KT / 32;
  __shared__ unsigned short Bs[128 * 40];           // padded stride: conflict-free

  const int tid  = threadIdx.x;
  const int lane = tid & 31;
  const int wave = tid >> 5;
  const int u0      = blockIdx.x * 32;
  const int rowBase = blockIdx.y * 64 + wave * 16;

  // B-load column for this thread: gate group (c>>5), unit u0+(c&31)
  const int c = tid;
  const int n = (c >> 5) * 512 + u0 + (c & 31);
  const unsigned short* Wrow = W + (size_t)n * KT;

  v8f acc[8];
#pragma unroll
  for (int i = 0; i < 8; ++i) acc[i] = (v8f)0.0f;

  const int arow  = rowBase + (lane & 15);
  const int khalf = lane >> 4;                      // 0 or 1

  for (int kc = 0; kc < NCH; ++kc) {
    const int kg = kc * 32;
    __syncthreads();
    {  // stage first 16 K-values of this thread's column chunk into LDS
      const U4* src = (const U4*)(Wrow + kg);
      U4* dst = (U4*)(&Bs[c * 40]);
      dst[0] = src[0];
      dst[1] = src[1];
      if (kg + 32 < KT) __builtin_prefetch(Wrow + kg + 32, 0, 0);
    }
    __syncthreads();

    // A fragment (wave32 WMMA swizzle): row = lane&15, K-half by lane>=16
    const unsigned short* Ap;
    int kk;
    if (kg < KA0) { Ap = A0 + (size_t)arow * sA0; kk = kg; }
    else          { Ap = A1 + (size_t)arow * 512; kk = kg - KA0; }
    const int kb = kk + khalf * 8;
    U8 au;
    au.lo = *(const U4*)(Ap + kb);
    au.hi = *(const U4*)(Ap + kb + 16);
    v16bf afrag = __builtin_bit_cast(v16bf, au);

    // Preload ALL B fragments, then fence the scheduler so no WMMA is
    // hoisted between the loads: forces 8 distinct fragment octets and
    // graduated s_wait_dscnt (loads return in order), so the 16
    // ds_load_b128 pipeline while independent WMMAs issue back-to-back.
    v16bf bfr[8];
#pragma unroll
    for (int nt = 0; nt < 8; ++nt) {
      const unsigned short* bp_ = &Bs[(nt * 16 + (lane & 15)) * 40 + khalf * 8];
      U8 bu;
      bu.lo = *(const U4*)bp_;
      bu.hi = *(const U4*)(bp_ + 16);
      bfr[nt] = __builtin_bit_cast(v16bf, bu);
    }
    __builtin_amdgcn_sched_barrier(0);              // loads above, WMMAs below
#pragma unroll
    for (int nt = 0; nt < 8; ++nt) {
      acc[nt] = __builtin_amdgcn_wmma_f32_16x16x32_bf16(
          false, afrag, false, bfr[nt], (short)0, acc[nt], false, false);
    }
  }

  // Epilogue: LSTM cell on registers.
  // C-tile layout: acc[nt][v] -> row = v + 8*(lane>=16), col = nt*16 + (lane&15)
  const int rl = khalf * 8;
#pragma unroll
  for (int v = 0; v < 8; ++v) {
    const int b = rowBase + v + rl;
    const bool m = t < lens[b];
#pragma unroll
    for (int p = 0; p < 2; ++p) {
      const int u = u0 + p * 16 + (lane & 15);
      float gi = acc[0 + p][v] + bias[u];
      float gj = acc[2 + p][v] + bias[512 + u];
      float gf = acc[4 + p][v] + bias[1024 + u];
      float go = acc[6 + p][v] + bias[1536 + u];
      float cold = cbuf[(size_t)b * 512 + u];
      float nc = cold * sigm(gf + 1.f) + sigm(gi) * tanh_fast(gj);
      float nh = tanh_fast(nc) * sigm(go);
      unsigned short nhb = f2bf(nh);
      if (hraw) hraw[(size_t)b * 512 + u] = nhb;              // raw (unmasked) feed
      unsigned short hold = A1[(size_t)b * 512 + u];          // previous state
      hstate[(size_t)b * 512 + u] = m ? nhb : hold;           // held state
      if (m) cbuf[(size_t)b * 512 + u] = nc;
      if (outs_t) outs_t[(size_t)b * 512 + u] = m ? nhb : (unsigned short)0;
    }
  }
}

// ---------------- projection: [65536][512] @ bf16[112][512]^T -> fp32 [B][T][100] ----
__global__ __launch_bounds__(128) void proj_kernel(
    const unsigned short* __restrict__ Hseq,   // [t*512+b][512] bf16
    const unsigned short* __restrict__ WpT,    // [112][512] bf16
    const float* __restrict__ bp,
    float* __restrict__ out) {
  __shared__ unsigned short Bs[112 * 40];
  const int tid  = threadIdx.x;
  const int lane = tid & 31;
  const int wave = tid >> 5;
  const int rowBase = blockIdx.x * 64 + wave * 16;

  v8f acc[7];
#pragma unroll
  for (int i = 0; i < 7; ++i) acc[i] = (v8f)0.0f;

  const int arow  = rowBase + (lane & 15);
  const int khalf = lane >> 4;

  for (int kc = 0; kc < 16; ++kc) {
    const int kg = kc * 32;
    __syncthreads();
    if (tid < 112) {
      const U4* src = (const U4*)(WpT + (size_t)tid * 512 + kg);
      U4* dst = (U4*)(&Bs[tid * 40]);
      dst[0] = src[0];
      dst[1] = src[1];
    }
    __syncthreads();

    const int kb = kg + khalf * 8;
    U8 au;
    au.lo = *(const U4*)(Hseq + (size_t)arow * 512 + kb);
    au.hi = *(const U4*)(Hseq + (size_t)arow * 512 + kb + 16);
    v16bf afrag = __builtin_bit_cast(v16bf, au);

    v16bf bfr[7];
#pragma unroll
    for (int nt = 0; nt < 7; ++nt) {
      const unsigned short* bp_ = &Bs[(nt * 16 + (lane & 15)) * 40 + khalf * 8];
      U8 bu;
      bu.lo = *(const U4*)bp_;
      bu.hi = *(const U4*)(bp_ + 16);
      bfr[nt] = __builtin_bit_cast(v16bf, bu);
    }
    __builtin_amdgcn_sched_barrier(0);              // loads above, WMMAs below
#pragma unroll
    for (int nt = 0; nt < 7; ++nt) {
      acc[nt] = __builtin_amdgcn_wmma_f32_16x16x32_bf16(
          false, afrag, false, bfr[nt], (short)0, acc[nt], false, false);
    }
  }

  const int rl = khalf * 8;
#pragma unroll
  for (int nt = 0; nt < 7; ++nt) {
#pragma unroll
    for (int v = 0; v < 8; ++v) {
      int r  = rowBase + v + rl;
      int tt = r >> 9, bb = r & 511;
      int vc = nt * 16 + (lane & 15);
      if (vc < V_SZ)
        out[(size_t)bb * (T_SZ * V_SZ) + (size_t)tt * V_SZ + vc] = acc[nt][v] + bp[vc];
    }
  }
}

// ---------------- host launcher ----------------
extern "C" void kernel_launch(void* const* d_in, const int* in_sizes, int n_in,
                              void* d_out, int out_size, void* d_ws, size_t ws_size,
                              hipStream_t stream) {
  const int*   dib   = (const int*)d_in[0];
  const int*   lens  = (const int*)d_in[1];
  const float* state = (const float*)d_in[2];
  const float* emb   = (const float*)d_in[3];
  const float* W0    = (const float*)d_in[4];
  const float* b0    = (const float*)d_in[5];
  const float* W1    = (const float*)d_in[6];
  const float* b1    = (const float*)d_in[7];
  const float* Wp    = (const float*)d_in[8];
  const float* bp    = (const float*)d_in[9];
  float*       out   = (float*)d_out;

  char* ws = (char*)d_ws;
  size_t off = 0;
  auto carve = [&](size_t bytes) -> char* {
    char* p = ws + off;
    off += (bytes + 255) & ~(size_t)255;
    return p;
  };
  unsigned short* Wpk0 = (unsigned short*)carve((size_t)2048 * 1088 * 2);
  unsigned short* Wpk1 = (unsigned short*)carve((size_t)2048 * 1024 * 2);
  unsigned short* WpkP = (unsigned short*)carve((size_t)112 * 512 * 2);
  unsigned short* Xall = (unsigned short*)carve((size_t)T_SZ * B_SZ * 576 * 2);
  unsigned short* outs = (unsigned short*)carve((size_t)T_SZ * B_SZ * 512 * 2);
  float*          c0   = (float*)carve((size_t)B_SZ * 512 * 4);
  float*          c1   = (float*)carve((size_t)B_SZ * 512 * 4);
  unsigned short* h0A  = (unsigned short*)carve((size_t)B_SZ * 512 * 2);
  unsigned short* h0B  = (unsigned short*)carve((size_t)B_SZ * 512 * 2);
  unsigned short* h1A  = (unsigned short*)carve((size_t)B_SZ * 512 * 2);
  unsigned short* h1B  = (unsigned short*)carve((size_t)B_SZ * 512 * 2);
  unsigned short* hraw = (unsigned short*)carve((size_t)B_SZ * 512 * 2);

  // prologue: pack weights (transpose to [N][K] bf16), build X, zero states
  pack_wT<<<dim3(1088 / 32, 2048 / 32), 256, 0, stream>>>(W0, Wpk0, 1088, 2048);
  pack_wT<<<dim3(1024 / 32, 2048 / 32), 256, 0, stream>>>(W1, Wpk1, 1024, 2048);
  pack_wp<<<(112 * 512 + 255) / 256, 256, 0, stream>>>(Wp, WpkP);
  {
    size_t total = (size_t)T_SZ * B_SZ * 576;
    build_x<<<(unsigned)((total + 255) / 256), 256, 0, stream>>>(dib, emb, state, Xall);
  }
  zero32<<<(262144 + 255) / 256, 256, 0, stream>>>((unsigned*)c0, 262144);
  zero32<<<(262144 + 255) / 256, 256, 0, stream>>>((unsigned*)c1, 262144);
  zero32<<<(131072 + 255) / 256, 256, 0, stream>>>((unsigned*)h0A, 131072);
  zero32<<<(131072 + 255) / 256, 256, 0, stream>>>((unsigned*)h1A, 131072);

  // recurrence: 128 steps, double-buffered h states
  for (int t = 0; t < T_SZ; ++t) {
    const unsigned short* h0r = (t & 1) ? h0B : h0A;
    unsigned short*       h0w = (t & 1) ? h0A : h0B;
    const unsigned short* h1r = (t & 1) ? h1B : h1A;
    unsigned short*       h1w = (t & 1) ? h1A : h1B;

    lstm_step_kernel<576, 512><<<dim3(16, 8), 128, 0, stream>>>(
        Xall + (size_t)t * B_SZ * 576, 576, h0r, Wpk0, b0,
        c0, hraw, h0w, nullptr, lens, t);

    lstm_step_kernel<512, 512><<<dim3(16, 8), 128, 0, stream>>>(
        hraw, 512, h1r, Wpk1, b1,
        c1, nullptr, h1w, outs + (size_t)t * B_SZ * 512, lens, t);
  }

  // final projection
  proj_kernel<<<(T_SZ * B_SZ) / 64, 128, 0, stream>>>(outs, WpkP, bp, out);
}
